// LSTM_single_23364622090949
// MI455X (gfx1250) — matrix operand order repaired and checked by reading, not compiled, not run on hardware
//
#include <hip/hip_runtime.h>
#include <hip/hip_bf16.h>
#include <stdint.h>

#define BB 64
#define TT 512
#define EE 1024
#define HH 1024
#define KA 2048   // H + E
#define N4 4096   // 4*H

#define AS1 __attribute__((address_space(1)))

typedef __bf16 bf16_t;
typedef __bf16 v16bf __attribute__((ext_vector_type(16)));
typedef __bf16 v8bf  __attribute__((ext_vector_type(8)));
typedef float v8f    __attribute__((ext_vector_type(8)));
typedef float v4f    __attribute__((ext_vector_type(4)));
typedef unsigned int v2u __attribute__((ext_vector_type(2)));

// ---- global-address-space load/store helpers (force global_load_*, not flat_load_*) ----
__device__ __forceinline__ v8bf   gld8(const bf16_t* p)  { return *(const v8bf  AS1*)(uintptr_t)p; }
__device__ __forceinline__ v16bf  gld16(const bf16_t* p) { return *(const v16bf AS1*)(uintptr_t)p; }
__device__ __forceinline__ float  gldf(const float* p)   { return *(const float AS1*)(uintptr_t)p; }
__device__ __forceinline__ v4f    gldf4(const float* p)  { return *(const v4f   AS1*)(uintptr_t)p; }
__device__ __forceinline__ void gstf(float* p, float v)      { *(float AS1*)(uintptr_t)p = v; }
__device__ __forceinline__ void gstb(bf16_t* p, bf16_t v)    { *(bf16_t AS1*)(uintptr_t)p = v; }
__device__ __forceinline__ void gstu2(void* p, v2u v)        { *(v2u AS1*)(uintptr_t)p = v; }

__device__ __forceinline__ float fsig(float x) {
    return 1.0f / (1.0f + __expf(-x));
}
__device__ __forceinline__ float ftanh(float x) {
    return 2.0f / (1.0f + __expf(-2.0f * x)) - 1.0f;
}

// ---------------- conversion kernels ----------------

// Wb layout: [4 gates][H rows][KA cols] bf16  (gate order: c, u, f, o)
__global__ void cvt_w_kernel(const float* __restrict__ wc, const float* __restrict__ wu,
                             const float* __restrict__ wf, const float* __restrict__ wo,
                             bf16_t* __restrict__ Wb) {
    const int g = blockIdx.y;
    const float* src = (g == 0) ? wc : (g == 1) ? wu : (g == 2) ? wf : wo;
    size_t i4 = ((size_t)blockIdx.x * blockDim.x + threadIdx.x) * 4;
    if (i4 >= (size_t)HH * KA) return;
    const v4f v = gldf4(src + i4);
    union { bf16_t b[4]; v2u u; } pk;
    pk.b[0] = (bf16_t)v[0]; pk.b[1] = (bf16_t)v[1];
    pk.b[2] = (bf16_t)v[2]; pk.b[3] = (bf16_t)v[3];
    gstu2(Wb + (size_t)g * HH * KA + i4, pk.u);
}

__global__ void cvt_x_kernel(const float* __restrict__ x, bf16_t* __restrict__ xb) {
    size_t i4 = ((size_t)blockIdx.x * blockDim.x + threadIdx.x) * 4;
    if (i4 >= (size_t)BB * TT * EE) return;
    const v4f v = gldf4(x + i4);
    union { bf16_t b[4]; v2u u; } pk;
    pk.b[0] = (bf16_t)v[0]; pk.b[1] = (bf16_t)v[1];
    pk.b[2] = (bf16_t)v[2]; pk.b[3] = (bf16_t)v[3];
    gstu2(xb + i4, pk.u);
}

__global__ void init_state_kernel(const float* __restrict__ a0, const float* __restrict__ c0,
                                  bf16_t* __restrict__ a_state, float* __restrict__ c_state) {
    size_t i = (size_t)blockIdx.x * blockDim.x + threadIdx.x;
    if (i >= (size_t)BB * HH) return;
    gstb(a_state + i, (bf16_t)gldf(a0 + i));
    gstf(c_state + i, gldf(c0 + i));
}

// ---------------- big parallel GEMM: input projections ----------------
// Zx[t][g][b][j] = bias[g][j] + sum_k x[b][t][k] * W[g][j][H + k]
// grid: (N4/128, (B*T)/128), block 256 threads = 8 waves (2 M x 4 N)
__global__ __launch_bounds__(256, 1)
void lstm_xproj_kernel(const bf16_t* __restrict__ xb, const bf16_t* __restrict__ Wb,
                       const float* __restrict__ bc, const float* __restrict__ bu,
                       const float* __restrict__ bfv, const float* __restrict__ bo,
                       float* __restrict__ Zx) {
    const int lane = threadIdx.x & 31;
    const int w    = threadIdx.x >> 5;
    const int mi = w & 1, ni = w >> 1;
    const int lhi = lane >> 4, llo = lane & 15;
    const int mbase = blockIdx.y * 128 + mi * 64;   // rows (b*T + t)
    const int nbase = blockIdx.x * 128 + ni * 32;   // cols (g*H + j)

    v8f acc[4][2];
#pragma unroll
    for (int nn = 0; nn < 2; ++nn) {
        const int n = nbase + nn * 16 + llo;
        const int g = n >> 10, j = n & 1023;
        const float* bp = (g == 0) ? bc : (g == 1) ? bu : (g == 2) ? bfv : bo;
        const float bias = gldf(bp + j);
#pragma unroll
        for (int mm = 0; mm < 4; ++mm)
#pragma unroll
            for (int v = 0; v < 8; ++v) acc[mm][nn][v] = bias;
    }

    const bf16_t* aptr[4];
#pragma unroll
    for (int mm = 0; mm < 4; ++mm)
        aptr[mm] = xb + (size_t)(mbase + mm * 16 + llo) * EE + lhi * 16;
    const bf16_t* bptr[2];
#pragma unroll
    for (int nn = 0; nn < 2; ++nn) {
        const int n = nbase + nn * 16 + llo;
        bptr[nn] = Wb + (size_t)n * KA + HH + lhi * 16;   // input-weight half
    }

    for (int k0 = 0; k0 < EE; k0 += 32) {
        v16bf afrag[4], bfrag[2];
#pragma unroll
        for (int mm = 0; mm < 4; ++mm) {
            v8bf lo = gld8(aptr[mm]);
            v8bf hi = gld8(aptr[mm] + 8);
#pragma unroll
            for (int e = 0; e < 8; ++e) { afrag[mm][e] = lo[e]; afrag[mm][e + 8] = hi[e]; }
            aptr[mm] += 32;
        }
#pragma unroll
        for (int nn = 0; nn < 2; ++nn) { bfrag[nn] = gld16(bptr[nn]); bptr[nn] += 32; }
#pragma unroll
        for (int mm = 0; mm < 4; ++mm)
#pragma unroll
            for (int nn = 0; nn < 2; ++nn)
                acc[mm][nn] = __builtin_amdgcn_wmma_f32_16x16x32_bf16(
                    false, afrag[mm], false, bfrag[nn], (short)0, acc[mm][nn], false, false);
    }

#pragma unroll
    for (int mm = 0; mm < 4; ++mm)
#pragma unroll
        for (int nn = 0; nn < 2; ++nn) {
            const int n = nbase + nn * 16 + llo;
            const int g = n >> 10, j = n & 1023;
#pragma unroll
            for (int v = 0; v < 8; ++v) {
                const int M = mbase + mm * 16 + v + lhi * 8;
                const int b = M >> 9, t = M & 511;   // row = b*T + t
                gstf(&Zx[(((size_t)t * 4 + g) * BB + b) * HH + j], acc[mm][nn][v]);
            }
        }
}

// ---------------- recurrent step kernel (launched once per t) ----------------
// grid 64 (j-tiles of 16), block 128 = 4 waves (one 16-row b-tile each).
// Each wave holds all four gates' accumulators -> fused LSTM update, no LDS.
__global__ __launch_bounds__(128, 1)
void lstm_step_kernel(const bf16_t* __restrict__ a_prev, bf16_t* __restrict__ a_next,
                      float* __restrict__ c_state,
                      const bf16_t* __restrict__ Wb, const float* __restrict__ Zx,
                      float* __restrict__ a_f, float* __restrict__ aT,
                      float* __restrict__ cT, int t) {
    const int lane = threadIdx.x & 31;
    const int w    = threadIdx.x >> 5;          // b-tile 0..3
    const int lhi = lane >> 4, llo = lane & 15;
    const int j = blockIdx.x * 16 + llo;        // column within gate

    v8f acc[4];
    const float* zbase = Zx + (size_t)t * 4 * BB * HH;
#pragma unroll
    for (int g = 0; g < 4; ++g)
#pragma unroll
        for (int v = 0; v < 8; ++v) {
            const int b = w * 16 + v + lhi * 8;
            acc[g][v] = gldf(&zbase[((size_t)g * BB + b) * HH + j]);
        }

    const bf16_t* ap = a_prev + (size_t)(w * 16 + llo) * HH + lhi * 16;
    const bf16_t* bp[4];
#pragma unroll
    for (int g = 0; g < 4; ++g)
        bp[g] = Wb + ((size_t)g * HH + j) * KA + lhi * 16;   // recurrent-weight half

#pragma unroll 2
    for (int k0 = 0; k0 < HH; k0 += 32) {
        v16bf afrag;
        v8bf lo = gld8(ap);
        v8bf hi = gld8(ap + 8);
#pragma unroll
        for (int e = 0; e < 8; ++e) { afrag[e] = lo[e]; afrag[e + 8] = hi[e]; }
        ap += 32;
#pragma unroll
        for (int g = 0; g < 4; ++g) {
            v16bf bfrag = gld16(bp[g]);
            bp[g] += 32;
            acc[g] = __builtin_amdgcn_wmma_f32_16x16x32_bf16(
                false, afrag, false, bfrag, (short)0, acc[g], false, false);
        }
    }

#pragma unroll
    for (int v = 0; v < 8; ++v) {
        const int b = w * 16 + v + lhi * 8;
        const size_t ix = (size_t)b * HH + j;
        const float cand = ftanh(acc[0][v]);
        const float gu   = fsig(acc[1][v]);
        const float gf   = fsig(acc[2][v]);
        const float go   = fsig(acc[3][v]);
        const float c1 = gu * cand + gf * gldf(&c_state[ix]);
        const float a1 = go * ftanh(c1);
        gstf(&c_state[ix], c1);
        gstb(&a_next[ix], (bf16_t)a1);
        gstf(&a_f[((size_t)b * TT + t) * HH + j], a1);
        if (t == TT - 1) { gstf(&aT[ix], a1); gstf(&cT[ix], c1); }
    }
}

// ---------------- host ----------------
extern "C" void kernel_launch(void* const* d_in, const int* in_sizes, int n_in,
                              void* d_out, int out_size, void* d_ws, size_t ws_size,
                              hipStream_t stream) {
    const float* x  = (const float*)d_in[0];
    const float* a0 = (const float*)d_in[1];
    const float* c0 = (const float*)d_in[2];
    const float* wc = (const float*)d_in[3];
    const float* wu = (const float*)d_in[4];
    const float* wf = (const float*)d_in[5];
    const float* wo = (const float*)d_in[6];
    const float* bc = (const float*)d_in[7];
    const float* bu = (const float*)d_in[8];
    const float* bfv = (const float*)d_in[9];
    const float* bo = (const float*)d_in[10];

    char* ws = (char*)d_ws;
    bf16_t* Wb = (bf16_t*)(ws);                         //  16 MB
    bf16_t* xb = (bf16_t*)(ws + (size_t)16777216);      //  64 MB
    float*  Zx = (float*)(ws + (size_t)83886080);       // 512 MB
    bf16_t* aA = (bf16_t*)(ws + (size_t)620756992);     // 128 KB
    bf16_t* aB = (bf16_t*)(ws + (size_t)620888064);     // 128 KB
    float*  cs = (float*)(ws + (size_t)621019136);      // 256 KB

    float* out = (float*)d_out;
    float* aT = out + (size_t)BB * TT * HH;
    float* cT = aT + (size_t)BB * HH;

    cvt_w_kernel<<<dim3((HH * KA / 4 + 255) / 256, 4), 256, 0, stream>>>(wc, wu, wf, wo, Wb);
    cvt_x_kernel<<<(BB * TT * EE / 4 + 255) / 256, 256, 0, stream>>>(x, xb);
    init_state_kernel<<<(BB * HH + 255) / 256, 256, 0, stream>>>(a0, c0, aA, cs);

    lstm_xproj_kernel<<<dim3(N4 / 128, (BB * TT) / 128), 256, 0, stream>>>(
        xb, Wb, bc, bu, bfv, bo, Zx);

    for (int t = 0; t < TT; ++t) {
        const bf16_t* ap = (t & 1) ? aB : aA;
        bf16_t*       an = (t & 1) ? aA : aB;
        lstm_step_kernel<<<HH / 16, 128, 0, stream>>>(ap, an, cs, Wb, Zx, out, aT, cT, t);
    }
}
